// TransformerModel_1271310319744
// MI455X (gfx1250) — matrix operand order repaired and checked by reading, not compiled, not run on hardware
//
#include <hip/hip_runtime.h>
#include <hip/hip_bf16.h>

// ---- model dims ----
#define B_  2
#define T_  1024
#define D_  1024
#define H_  16
#define HD_ 64
#define FF_ 4096
#define L_  6
#define V_  50257
#define M_  (B_ * T_)

typedef __attribute__((ext_vector_type(16))) _Float16 v16h;
typedef __attribute__((ext_vector_type(8)))  float    v8f;
typedef __attribute__((ext_vector_type(4)))  _Float16 v4h;

union F16Frag { v16h v; uint4 q[2]; v4h h4[4]; };

// ======================= WMMA GEMM =======================
// C[M,N] = A[M,K] @ W[K,N] + bias[N] (+res[M,N]) (relu?)
// 256 threads (8 waves), tile 128x128, K-step 32, double-buffered LDS,
// software-pipelined global->reg->LDS staging (f32 -> f16 convert on the fly).
// B staging: 4x4 register micro-transpose -> 4x ds_store_b64 per thread.
#define TILE_M 128
#define TILE_N 128
#define TILE_K 32
#define LDA 40   // halves per As row (32 + 8 pad); rows stay 16B aligned
#define LDB 40   // halves per Bs row (K-major, transposed W)

template <bool NGUARD>
__global__ __launch_bounds__(256)
void gemm_wmma_f16(const float* __restrict__ A, const float* __restrict__ W,
                   const float* __restrict__ bias, const float* __restrict__ res,
                   float* __restrict__ C, int Ndim, int Kdim, int relu)
{
    __shared__ __align__(16) _Float16 As[2][TILE_M * LDA];
    __shared__ __align__(16) _Float16 Bs[2][TILE_N * LDB];

    const int t  = threadIdx.x;
    const int m0 = blockIdx.y * TILE_M;
    const int n0 = blockIdx.x * TILE_N;

    const int lane = t & 31;
    const int wid  = t >> 5;
    const int wm   = (wid & 1) * 64;   // 2 waves along M
    const int wn   = (wid >> 1) * 32;  // 4 waves along N
    const int lr   = lane & 15;
    const int hf   = lane >> 4;        // half-wave select

    // loop-invariant staging coordinates
    int rowA[4], colA[4];
    #pragma unroll
    for (int i = 0; i < 4; ++i) {
        int j   = t + i * 256;
        rowA[i] = j >> 3;              // 8 float4 per 32-float A row
        colA[i] = (j & 7) << 2;
    }
    const int nB = (t & 31) << 2;      // column group (4 cols) of W tile
    const int kB = (t >> 5) << 2;      // row group (4 rows) of W tile

    float4 ar[4], br[4];               // in-flight staging registers

    auto loadA = [&](int k0) {
        #pragma unroll
        for (int i = 0; i < 4; ++i)
            ar[i] = *reinterpret_cast<const float4*>(
                A + (size_t)(m0 + rowA[i]) * Kdim + k0 + colA[i]);
    };
    auto loadB = [&](int k0) {
        int gn = n0 + nB;
        if (!NGUARD || gn + 3 < Ndim) {
            #pragma unroll
            for (int i = 0; i < 4; ++i)
                br[i] = *reinterpret_cast<const float4*>(
                    W + (size_t)(k0 + kB + i) * Ndim + gn);
        } else {
            #pragma unroll
            for (int i = 0; i < 4; ++i) {
                const float* wp = W + (size_t)(k0 + kB + i) * Ndim + gn;
                #pragma unroll
                for (int c = 0; c < 4; ++c)
                    reinterpret_cast<float*>(&br[i])[c] = (gn + c < Ndim) ? wp[c] : 0.f;
            }
        }
    };
    auto storeA = [&](int buf) {
        #pragma unroll
        for (int i = 0; i < 4; ++i) {
            v4h hv = { (_Float16)ar[i].x, (_Float16)ar[i].y,
                       (_Float16)ar[i].z, (_Float16)ar[i].w };
            *reinterpret_cast<v4h*>(&As[buf][rowA[i] * LDA + colA[i]]) = hv;
        }
    };
    auto storeB = [&](int buf) {       // 4x4 register transpose -> Bs[n][k]
        #pragma unroll
        for (int c = 0; c < 4; ++c) {
            v4h hv = { (_Float16)reinterpret_cast<const float*>(&br[0])[c],
                       (_Float16)reinterpret_cast<const float*>(&br[1])[c],
                       (_Float16)reinterpret_cast<const float*>(&br[2])[c],
                       (_Float16)reinterpret_cast<const float*>(&br[3])[c] };
            *reinterpret_cast<v4h*>(&Bs[buf][(nB + c) * LDB + kB]) = hv;
        }
    };

    v8f acc[4][2];
    #pragma unroll
    for (int i = 0; i < 4; ++i)
        #pragma unroll
        for (int j = 0; j < 2; ++j)
            #pragma unroll
            for (int r = 0; r < 8; ++r)
                acc[i][j][r] = 0.0f;

    const int nk = Kdim / TILE_K;

    // prologue: tile 0 -> LDS buf0; tile 1 -> regs
    loadA(0); loadB(0);
    storeA(0); storeB(0);
    if (nk > 1) { loadA(TILE_K); loadB(TILE_K); }
    __syncthreads();

    for (int ki = 0; ki < nk; ++ki) {
        const int cur = ki & 1;
        if (ki + 1 < nk) { storeA(cur ^ 1); storeB(cur ^ 1); }
        if (ki + 2 < nk) { loadA((ki + 2) * TILE_K); loadB((ki + 2) * TILE_K); }

        F16Frag a[4], b[2];
        #pragma unroll
        for (int i = 0; i < 4; ++i) {
            const _Float16* p = &As[cur][(wm + i * 16 + lr) * LDA + hf * 8];
            a[i].q[0] = *reinterpret_cast<const uint4*>(p);
            a[i].q[1] = *reinterpret_cast<const uint4*>(p + 16);
        }
        #pragma unroll
        for (int j = 0; j < 2; ++j) {
            const _Float16* p = &Bs[cur][(wn + j * 16 + lr) * LDB + hf * 8];
            b[j].q[0] = *reinterpret_cast<const uint4*>(p);
            b[j].q[1] = *reinterpret_cast<const uint4*>(p + 16);
        }
        #pragma unroll
        for (int i = 0; i < 4; ++i)
            #pragma unroll
            for (int j = 0; j < 2; ++j)
                acc[i][j] = __builtin_amdgcn_wmma_f32_16x16x32_f16(
                    false, a[i].v, false, b[j].v, (short)0, acc[i][j], false, false);

        __syncthreads();   // single barrier per K-step (ping-pong buffers)
    }

    // ---- epilogue: bias / residual / relu; C layout: VGPR r -> row hf*8+r ----
    #pragma unroll
    for (int i = 0; i < 4; ++i) {
        #pragma unroll
        for (int j = 0; j < 2; ++j) {
            int col = n0 + wn + j * 16 + lr;
            if (!NGUARD || col < Ndim) {
                float bb = bias[col];
                #pragma unroll
                for (int r = 0; r < 8; ++r) {
                    int row = m0 + wm + i * 16 + hf * 8 + r;
                    float vv = acc[i][j][r] + bb;
                    if (res)  vv += res[(size_t)row * Ndim + col];
                    if (relu) vv  = fmaxf(vv, 0.0f);
                    C[(size_t)row * Ndim + col] = vv;
                }
            }
        }
    }
}

// ======================= LayerNorm =======================
__global__ __launch_bounds__(256)
void ln_kernel(const float* __restrict__ x, const float* __restrict__ w,
               const float* __restrict__ b, float* __restrict__ y)
{
    const int row = blockIdx.x;
    const float* xr = x + (size_t)row * D_;
    float s = 0.f, s2 = 0.f;
    float vals[4];
    #pragma unroll
    for (int i = 0; i < 4; ++i) {
        float v = xr[threadIdx.x + i * 256];
        vals[i] = v; s += v; s2 += v * v;
    }
    #pragma unroll
    for (int off = 16; off > 0; off >>= 1) {   // wave32 shuffle reduce
        s  += __shfl_down(s,  off);
        s2 += __shfl_down(s2, off);
    }
    __shared__ float rs[8], rs2[8];
    __shared__ float mu_s, rstd_s;
    const int lane = threadIdx.x & 31, wid = threadIdx.x >> 5;
    if (lane == 0) { rs[wid] = s; rs2[wid] = s2; }
    __syncthreads();
    if (threadIdx.x == 0) {
        float ts = 0.f, ts2 = 0.f;
        #pragma unroll
        for (int i = 0; i < 8; ++i) { ts += rs[i]; ts2 += rs2[i]; }
        float mu  = ts * (1.0f / D_);
        float var = ts2 * (1.0f / D_) - mu * mu;
        mu_s = mu; rstd_s = rsqrtf(var + 1e-5f);
    }
    __syncthreads();
    const float mu = mu_s, rstd = rstd_s;
    #pragma unroll
    for (int i = 0; i < 4; ++i) {
        int c = threadIdx.x + i * 256;
        y[(size_t)row * D_ + c] = (vals[i] - mu) * rstd * w[c] + b[c];
    }
}

// ======================= Embedding =======================
__global__ __launch_bounds__(256)
void embed_kernel(const int* __restrict__ x, const float* __restrict__ tok,
                  const float* __restrict__ pos, float* __restrict__ h)
{
    int e4  = blockIdx.x * 256 + threadIdx.x;  // float4 index, total M_*D_/4
    int row = e4 >> 8;                          // 256 float4 per row
    int c   = (e4 & 255) << 2;
    int token = x[row];
    int tpos  = row & (T_ - 1);
    float4 a = *reinterpret_cast<const float4*>(tok + (size_t)token * D_ + c);
    float4 p = *reinterpret_cast<const float4*>(pos + (size_t)tpos  * D_ + c);
    *reinterpret_cast<float4*>(h + (size_t)row * D_ + c) =
        make_float4(a.x + p.x, a.y + p.y, a.z + p.z, a.w + p.w);
}

// ======================= WMMA flash attention =======================
// grid (T/128, B*H); 8 waves x 16 queries; key chunks of 32 staged in LDS f16.
// S = Q K^T via 4 wmma, online softmax on C-fragment layout, O += P V via 4 wmma.
// Interior chunks (kc0+31 <= q0) take a mask-free fast path (wave-uniform).
#define LDK 72   // Ks row stride (64 + 8 halves), 144B: 16B aligned
#define LDV 40   // Vt row stride (32 + 8 halves), 80B: 16B aligned
#define LDP 32   // Ps row stride (halves)

__global__ __launch_bounds__(256)
void flash_attn_kernel(const float* __restrict__ qg, const float* __restrict__ kg,
                       const float* __restrict__ vg, float* __restrict__ og)
{
    __shared__ __align__(16) _Float16 Ks[32 * LDK];      // [key][d]
    __shared__ __align__(16) _Float16 Vt[HD_ * LDV];     // [d][key]
    __shared__ __align__(16) _Float16 Ps[8][16 * LDP];   // per-wave P tile

    const int t    = threadIdx.x;
    const int lane = t & 31;
    const int w    = t >> 5;
    const int lr   = lane & 15;
    const int hf   = lane >> 4;

    const int bh = blockIdx.y;
    const int b  = bh >> 4;
    const int h  = bh & (H_ - 1);
    const size_t head = ((size_t)b * T_) * (H_ * HD_) + (size_t)h * HD_;
    const size_t rs   = H_ * HD_;

    const int q0 = blockIdx.x * 128 + w * 16;

    // ---- Q A-fragments (dc = 0, 32), resident in registers ----
    F16Frag qa[2];
    {
        const float* qrow = qg + head + (size_t)(q0 + lr) * rs;
        #pragma unroll
        for (int f = 0; f < 2; ++f) {
            int b1 = f * 32 + hf * 16;       // halves 0..7  (d)
            int b2 = f * 32 + hf * 16 + 8;   // halves 8..15 (d)
            float4 x0 = *reinterpret_cast<const float4*>(qrow + b1);
            float4 x1 = *reinterpret_cast<const float4*>(qrow + b1 + 4);
            float4 x2 = *reinterpret_cast<const float4*>(qrow + b2);
            float4 x3 = *reinterpret_cast<const float4*>(qrow + b2 + 4);
            qa[f].h4[0] = v4h{(_Float16)x0.x,(_Float16)x0.y,(_Float16)x0.z,(_Float16)x0.w};
            qa[f].h4[1] = v4h{(_Float16)x1.x,(_Float16)x1.y,(_Float16)x1.z,(_Float16)x1.w};
            qa[f].h4[2] = v4h{(_Float16)x2.x,(_Float16)x2.y,(_Float16)x2.z,(_Float16)x2.w};
            qa[f].h4[3] = v4h{(_Float16)x3.x,(_Float16)x3.y,(_Float16)x3.z,(_Float16)x3.w};
        }
    }

    v8f ov[4];
    #pragma unroll
    for (int nt = 0; nt < 4; ++nt)
        #pragma unroll
        for (int r = 0; r < 8; ++r)
            ov[nt][r] = 0.0f;
    float mrow[8], ssum[8];
    #pragma unroll
    for (int r = 0; r < 8; ++r) { mrow[r] = -3.0e38f; ssum[r] = 0.f; }

    const int nch = blockIdx.x * 4 + 4;     // chunks of 32 keys this block needs
    for (int c = 0; c < nch; ++c) {
        const int kc0 = c * 32;
        // ---- cooperative staging (all 8 waves) ----
        #pragma unroll
        for (int i = 0; i < 2; ++i) {        // Ks: 32x64 f32 -> f16, direct
            int j   = t + i * 256;           // 512 float4
            int key = j >> 4;
            int d4  = (j & 15) << 2;
            float4 kv = *reinterpret_cast<const float4*>(
                kg + head + (size_t)(kc0 + key) * rs + d4);
            *reinterpret_cast<v4h*>(&Ks[key * LDK + d4]) =
                v4h{(_Float16)kv.x,(_Float16)kv.y,(_Float16)kv.z,(_Float16)kv.w};
        }
        if (t < 128) {                       // Vt: 32x64 -> [d][key] via 4x4 transpose
            int kb = (t & 7) << 2;
            int db = (t >> 3) << 2;
            float4 vr[4];
            #pragma unroll
            for (int i = 0; i < 4; ++i)
                vr[i] = *reinterpret_cast<const float4*>(
                    vg + head + (size_t)(kc0 + kb + i) * rs + db);
            #pragma unroll
            for (int cc = 0; cc < 4; ++cc) {
                v4h hv = { (_Float16)reinterpret_cast<const float*>(&vr[0])[cc],
                           (_Float16)reinterpret_cast<const float*>(&vr[1])[cc],
                           (_Float16)reinterpret_cast<const float*>(&vr[2])[cc],
                           (_Float16)reinterpret_cast<const float*>(&vr[3])[cc] };
                *reinterpret_cast<v4h*>(&Vt[(db + cc) * LDV + kb]) = hv;
            }
        }
        __syncthreads();

        if (kc0 <= q0 + 15) {                // wave-uniform: EXEC stays all-ones
            // ---- S = Q K^T : two 16x16 key tiles, K-dim 64 split in two ----
            v8f s0, s1;
            #pragma unroll
            for (int r = 0; r < 8; ++r) { s0[r] = 0.f; s1[r] = 0.f; }
            #pragma unroll
            for (int f = 0; f < 2; ++f) {
                int dc = f * 32;
                F16Frag kb0, kb1;
                const _Float16* p0 = &Ks[lr * LDK + dc + hf * 16];
                kb0.q[0] = *reinterpret_cast<const uint4*>(p0);
                kb0.q[1] = *reinterpret_cast<const uint4*>(p0 + 8);
                const _Float16* p1 = &Ks[(16 + lr) * LDK + dc + hf * 16];
                kb1.q[0] = *reinterpret_cast<const uint4*>(p1);
                kb1.q[1] = *reinterpret_cast<const uint4*>(p1 + 8);
                s0 = __builtin_amdgcn_wmma_f32_16x16x32_f16(
                    false, qa[f].v, false, kb0.v, (short)0, s0, false, false);
                s1 = __builtin_amdgcn_wmma_f32_16x16x32_f16(
                    false, qa[f].v, false, kb1.v, (short)0, s1, false, false);
            }
            // ---- scale (+ causal mask only on the diagonal chunk) + online softmax ----
            const bool diag = (kc0 + 31 > q0);   // wave-uniform
            float p0a[8], p1a[8];
            #pragma unroll
            for (int r = 0; r < 8; ++r) {
                float sc0 = s0[r] * 0.125f;
                float sc1 = s1[r] * 0.125f;
                if (diag) {                      // only ~1 chunk per wave pays this
                    int qq = q0 + hf * 8 + r;
                    if (kc0 + lr      > qq) sc0 = -3.0e38f;
                    if (kc0 + 16 + lr > qq) sc1 = -3.0e38f;
                }
                float cm = fmaxf(sc0, sc1);
                #pragma unroll
                for (int mk = 1; mk <= 8; mk <<= 1) cm = fmaxf(cm, __shfl_xor(cm, mk));
                float nm   = fmaxf(mrow[r], cm);
                float wold = __expf(mrow[r] - nm);
                mrow[r] = nm;
                float e0 = __expf(sc0 - nm);
                float e1 = __expf(sc1 - nm);
                float ps = e0 + e1;
                #pragma unroll
                for (int mk = 1; mk <= 8; mk <<= 1) ps += __shfl_xor(ps, mk);
                ssum[r] = ssum[r] * wold + ps;
                #pragma unroll
                for (int nt = 0; nt < 4; ++nt) ov[nt][r] *= wold;
                p0a[r] = e0; p1a[r] = e1;
            }
            // ---- P: C-layout -> A-layout through per-wave LDS (intra-wave order) ----
            #pragma unroll
            for (int r = 0; r < 8; ++r) {
                Ps[w][(hf * 8 + r) * LDP + lr]      = (_Float16)p0a[r];
                Ps[w][(hf * 8 + r) * LDP + 16 + lr] = (_Float16)p1a[r];
            }
            F16Frag pa;
            const _Float16* pp = &Ps[w][lr * LDP + hf * 8];
            pa.q[0] = *reinterpret_cast<const uint4*>(pp);
            pa.q[1] = *reinterpret_cast<const uint4*>(pp + 16);
            // ---- O += P V ----
            #pragma unroll
            for (int nt = 0; nt < 4; ++nt) {
                F16Frag vb;
                const _Float16* vp = &Vt[(nt * 16 + lr) * LDV + hf * 8];
                vb.q[0] = *reinterpret_cast<const uint4*>(vp);
                vb.q[1] = *reinterpret_cast<const uint4*>(vp + 16);
                ov[nt] = __builtin_amdgcn_wmma_f32_16x16x32_f16(
                    false, pa.v, false, vb.v, (short)0, ov[nt], false, false);
            }
        }
        __syncthreads();
    }

    // ---- epilogue: O / rowsum -> att[b,t,h,d] ----
    #pragma unroll
    for (int r = 0; r < 8; ++r) {
        float inv = 1.f / ssum[r];
        int qq = q0 + hf * 8 + r;
        float* orow = og + head + (size_t)qq * rs;
        #pragma unroll
        for (int nt = 0; nt < 4; ++nt)
            orow[nt * 16 + lr] = ov[nt][r] * inv;
    }
}

// ======================= Orchestration =======================
extern "C" void kernel_launch(void* const* d_in, const int* in_sizes, int n_in,
                              void* d_out, int out_size, void* d_ws, size_t ws_size,
                              hipStream_t stream)
{
    (void)in_sizes; (void)n_in; (void)out_size; (void)ws_size;

    const int*   x       = (const int*)  d_in[0];
    const float* tok_emb = (const float*)d_in[1];
    const float* pos_emb = (const float*)d_in[2];
    const float* wq      = (const float*)d_in[3];
    const float* bq      = (const float*)d_in[4];
    const float* wk      = (const float*)d_in[5];
    const float* bk      = (const float*)d_in[6];
    const float* wv      = (const float*)d_in[7];
    const float* bv      = (const float*)d_in[8];
    const float* wo      = (const float*)d_in[9];
    const float* bo      = (const float*)d_in[10];
    const float* ln1_w   = (const float*)d_in[11];
    const float* ln1_b   = (const float*)d_in[12];
    const float* ln2_w   = (const float*)d_in[13];
    const float* ln2_b   = (const float*)d_in[14];
    const float* ffn_w1  = (const float*)d_in[15];
    const float* ffn_b1  = (const float*)d_in[16];
    const float* ffn_w2  = (const float*)d_in[17];
    const float* ffn_b2  = (const float*)d_in[18];
    const float* lnf_w   = (const float*)d_in[19];
    const float* lnf_b   = (const float*)d_in[20];
    const float* proj_w  = (const float*)d_in[21];
    const float* proj_b  = (const float*)d_in[22];

    float* ws = (float*)d_ws;
    const size_t S = (size_t)M_ * D_;
    float* h   = ws;
    float* tmp = ws + 1 * S;
    float* q   = ws + 2 * S;
    float* kk  = ws + 3 * S;
    float* vv  = ws + 4 * S;
    float* att = ws + 5 * S;
    float* ffh = ws + 6 * S;   // M_ x FF_ (4*S)

    const dim3 blk(256);
    const dim3 gDD(D_  / TILE_N, M_ / TILE_M);
    const dim3 gDF(FF_ / TILE_N, M_ / TILE_M);
    const dim3 gDV((V_ + TILE_N - 1) / TILE_N, M_ / TILE_M);
    const dim3 gFA(T_ / 128, B_ * H_);

    embed_kernel<<<(M_ * D_ / 4) / 256, blk, 0, stream>>>(x, tok_emb, pos_emb, h);

    for (int l = 0; l < L_; ++l) {
        ln_kernel<<<M_, blk, 0, stream>>>(h, ln1_w + l * D_, ln1_b + l * D_, tmp);
        gemm_wmma_f16<false><<<gDD, blk, 0, stream>>>(tmp, wq + (size_t)l * D_ * D_, bq + l * D_,
                                                      nullptr, q,  D_, D_, 0);
        gemm_wmma_f16<false><<<gDD, blk, 0, stream>>>(tmp, wk + (size_t)l * D_ * D_, bk + l * D_,
                                                      nullptr, kk, D_, D_, 0);
        gemm_wmma_f16<false><<<gDD, blk, 0, stream>>>(tmp, wv + (size_t)l * D_ * D_, bv + l * D_,
                                                      nullptr, vv, D_, D_, 0);
        flash_attn_kernel<<<gFA, blk, 0, stream>>>(q, kk, vv, att);
        gemm_wmma_f16<false><<<gDD, blk, 0, stream>>>(att, wo + (size_t)l * D_ * D_, bo + l * D_,
                                                      h, h, D_, D_, 0);        // residual fused
        ln_kernel<<<M_, blk, 0, stream>>>(h, ln2_w + l * D_, ln2_b + l * D_, tmp);
        gemm_wmma_f16<false><<<gDF, blk, 0, stream>>>(tmp, ffn_w1 + (size_t)l * D_ * FF_, ffn_b1 + l * FF_,
                                                      nullptr, ffh, FF_, D_, 1);   // relu fused
        gemm_wmma_f16<false><<<gDD, blk, 0, stream>>>(ffh, ffn_w2 + (size_t)l * FF_ * D_, ffn_b2 + l * D_,
                                                      h, h, D_, FF_, 0);       // residual fused
    }

    ln_kernel<<<M_, blk, 0, stream>>>(h, lnf_w, lnf_b, tmp);
    gemm_wmma_f16<true><<<gDV, blk, 0, stream>>>(tmp, proj_w, proj_b, nullptr,
                                                 (float*)d_out, V_, D_, 0);
}
